// Model_39676907886676
// MI455X (gfx1250) — matrix operand order repaired and checked by reading, not compiled, not run on hardware
//
#include <hip/hip_runtime.h>

// ---------------------------------------------------------------------------
// Flash attention, B=2, S=8192, D=64, fp32 in/out, scale 1/8.
// bf16 WMMA (v_wmma_f32_16x16x32_bf16), f32 accumulate, online softmax.
// KT=64 K-columns/iter -> 16 WMMA/iter. All-asm softmax row head
// (v_max3_num_f32 + v_max_num_f32 + v_max_num_f32_dpp row_xmask), packed
// fp32 body via v_pk_add/mul/fma_f32, exp2-domain softmax, v_rcp epilogue,
// packed bf16 P stores with k-interleave shared by the V staging.
// ---------------------------------------------------------------------------

typedef __attribute__((ext_vector_type(16))) __bf16 bf16x16;
typedef __attribute__((ext_vector_type(8)))  __bf16 bf16x8;
typedef __attribute__((ext_vector_type(2)))  __bf16 bf16x2;
typedef __attribute__((ext_vector_type(8)))  float  floatx8;
typedef __attribute__((ext_vector_type(2)))  float  floatx2;

#define S_LEN 8192
#define D_DIM 64
#define KT    64
#define QW    16
#define WAVES 8
#define KSTR  72
#define VSTR  72
#define PSTR  72

// log2(e)/8 folded into Q: softmax runs in the exp2 domain.
#define QSCALE (0.125f * 1.44269504088896340736f)

__device__ __forceinline__ bf16x16 lds_ld32(const __bf16* p) {
  bf16x8 a = *(const bf16x8*)p;
  bf16x8 b = *(const bf16x8*)(p + 8);
  bf16x16 r;
#pragma unroll
  for (int j = 0; j < 8; ++j) { r[j] = a[j]; r[8 + j] = b[j]; }
  return r;
}

// ---- asm float max helpers (no canonicalize insertion) ---------------------
__device__ __forceinline__ float vmax(float a, float b) {
  float d; asm("v_max_num_f32 %0, %1, %2" : "=v"(d) : "v"(a), "v"(b)); return d;
}
__device__ __forceinline__ float vmax3(float a, float b, float c) {
  float d; asm("v_max3_num_f32 %0, %1, %2, %3" : "=v"(d) : "v"(a), "v"(b), "v"(c)); return d;
}

// ---- packed fp32 VOP3P helpers --------------------------------------------
__device__ __forceinline__ floatx2 pk_add(floatx2 a, floatx2 b) {
  floatx2 d; asm("v_pk_add_f32 %0, %1, %2" : "=v"(d) : "v"(a), "v"(b)); return d;
}
__device__ __forceinline__ floatx2 pk_sub(floatx2 a, floatx2 b) {
  floatx2 d;
  asm("v_pk_add_f32 %0, %1, %2 neg_lo:[0,1] neg_hi:[0,1]" : "=v"(d) : "v"(a), "v"(b));
  return d;
}
__device__ __forceinline__ floatx2 pk_mul(floatx2 a, floatx2 b) {
  floatx2 d; asm("v_pk_mul_f32 %0, %1, %2" : "=v"(d) : "v"(a), "v"(b)); return d;
}
__device__ __forceinline__ floatx2 pk_fma(floatx2 a, floatx2 b, floatx2 c) {
  floatx2 d; asm("v_pk_fma_f32 %0, %1, %2, %3" : "=v"(d) : "v"(a), "v"(b), "v"(c)); return d;
}

// One-instruction-per-step 16-lane row reductions: DPP on src0 of VOP2.
__device__ __forceinline__ float rowmax16(float v) {
  float a, b, c, d;
  asm("v_max_num_f32_dpp %0, %1, %1 row_xmask:1 row_mask:0xf bank_mask:0xf bound_ctrl:1"
      : "=&v"(a) : "v"(v));
  asm("v_max_num_f32_dpp %0, %1, %1 row_xmask:2 row_mask:0xf bank_mask:0xf bound_ctrl:1"
      : "=&v"(b) : "v"(a));
  asm("v_max_num_f32_dpp %0, %1, %1 row_xmask:4 row_mask:0xf bank_mask:0xf bound_ctrl:1"
      : "=&v"(c) : "v"(b));
  asm("v_max_num_f32_dpp %0, %1, %1 row_xmask:8 row_mask:0xf bank_mask:0xf bound_ctrl:1"
      : "=&v"(d) : "v"(c));
  return d;
}
__device__ __forceinline__ float rowsum16(float v) {
  float a, b, c, d;
  asm("v_add_f32_dpp %0, %1, %1 row_xmask:1 row_mask:0xf bank_mask:0xf bound_ctrl:1"
      : "=&v"(a) : "v"(v));
  asm("v_add_f32_dpp %0, %1, %1 row_xmask:2 row_mask:0xf bank_mask:0xf bound_ctrl:1"
      : "=&v"(b) : "v"(a));
  asm("v_add_f32_dpp %0, %1, %1 row_xmask:4 row_mask:0xf bank_mask:0xf bound_ctrl:1"
      : "=&v"(c) : "v"(b));
  asm("v_add_f32_dpp %0, %1, %1 row_xmask:8 row_mask:0xf bank_mask:0xf bound_ctrl:1"
      : "=&v"(d) : "v"(c));
  return d;
}

__global__ __launch_bounds__(256, 1)
void fa_fwd_kernel(const float* __restrict__ Q,
                   const float* __restrict__ K,
                   const float* __restrict__ V,
                   float* __restrict__ O) {
  __shared__ __attribute__((aligned(16))) __bf16 Ksh[KT * KSTR];         // [k][d]
  __shared__ __attribute__((aligned(16))) __bf16 VshT[D_DIM * VSTR];     // [d][kperm]
  __shared__ __attribute__((aligned(16))) __bf16 Psc[WAVES][QW * PSTR];  // [q][kperm]

  const int tid  = threadIdx.x;
  const int wave = tid >> 5;
  const int lane = tid & 31;
  const int lm   = lane & 15;
  const int hi   = lane >> 4;

  const int blk  = blockIdx.x;     // 0..127
  const int b    = blk >> 6;
  const int qblk = blk & 63;
  const int q0   = qblk * 128 + wave * QW;

  const float* Qb = Q + (long)b * S_LEN * D_DIM;
  const float* Kb = K + (long)b * S_LEN * D_DIM;
  const float* Vb = V + (long)b * S_LEN * D_DIM;
  float*       Ob = O + (long)b * S_LEN * D_DIM;

  // ---- Q -> A-operand registers (pre-scaled into log2 domain) -------------
  bf16x16 aQ0, aQ1;
  {
    const float* qp = Qb + (long)(q0 + lm) * D_DIM;
#pragma unroll
    for (int j = 0; j < 8; ++j) {
      aQ0[j]     = (__bf16)(qp[     hi * 16 + j] * QSCALE);
      aQ0[8 + j] = (__bf16)(qp[ 8 + hi * 16 + j] * QSCALE);
      aQ1[j]     = (__bf16)(qp[32 + hi * 16 + j] * QSCALE);
      aQ1[8 + j] = (__bf16)(qp[40 + hi * 16 + j] * QSCALE);
    }
  }

  float rm[8];                 // row max (replicated in 16-lane row)
  floatx8 rlv = {};            // per-lane PARTIAL row sums
  floatx8 o0 = {}, o1 = {}, o2 = {}, o3 = {};
#pragma unroll
  for (int r = 0; r < 8; ++r) rm[r] = -1e30f;

  __bf16* Pw = &Psc[wave][0];

  for (int kb = 0; kb < S_LEN; kb += KT) {
    // ---- cooperative K/V tile load: 64x64 f32 -> bf16 LDS -----------------
    {
      const int row = tid >> 2;            // 0..63
      const int c16 = (tid & 3) * 16;      // 0,16,32,48
      const float* kp = Kb + (long)(kb + row) * D_DIM + c16;
      const float* vp = Vb + (long)(kb + row) * D_DIM + c16;
      bf16x8 ka, kc;
#pragma unroll
      for (int j = 0; j < 8; ++j) { ka[j] = (__bf16)kp[j]; kc[j] = (__bf16)kp[8 + j]; }
      *(bf16x8*)&Ksh[row * KSTR + c16]     = ka;
      *(bf16x8*)&Ksh[row * KSTR + c16 + 8] = kc;
      // V transposed + k-interleave (matches packed P stores):
      const int pcol = (row & 32) + 2 * (row & 15) + ((row >> 4) & 1);
#pragma unroll
      for (int j = 0; j < 16; ++j)
        VshT[(c16 + j) * VSTR + pcol] = (__bf16)vp[j];
      if (kb + KT < S_LEN) {
        __builtin_prefetch(kp + KT * D_DIM, 0, 0);
        __builtin_prefetch(vp + KT * D_DIM, 0, 0);
      }
    }
    __syncthreads();

    // ---- scores: four 16x16 f32 tiles -------------------------------------
    floatx8 z = {};
    floatx8 s0, s1, s2, s3;
    {
      const __bf16* kr0 = &Ksh[lm * KSTR];
      const __bf16* kr1 = &Ksh[(16 + lm) * KSTR];
      const __bf16* kr2 = &Ksh[(32 + lm) * KSTR];
      const __bf16* kr3 = &Ksh[(48 + lm) * KSTR];
      s0 = __builtin_amdgcn_wmma_f32_16x16x32_bf16(false, aQ0, false, lds_ld32(kr0 + hi * 16), (short)0, z, false, false);
      s1 = __builtin_amdgcn_wmma_f32_16x16x32_bf16(false, aQ0, false, lds_ld32(kr1 + hi * 16), (short)0, z, false, false);
      s2 = __builtin_amdgcn_wmma_f32_16x16x32_bf16(false, aQ0, false, lds_ld32(kr2 + hi * 16), (short)0, z, false, false);
      s3 = __builtin_amdgcn_wmma_f32_16x16x32_bf16(false, aQ0, false, lds_ld32(kr3 + hi * 16), (short)0, z, false, false);
      s0 = __builtin_amdgcn_wmma_f32_16x16x32_bf16(false, aQ1, false, lds_ld32(kr0 + 32 + hi * 16), (short)0, s0, false, false);
      s1 = __builtin_amdgcn_wmma_f32_16x16x32_bf16(false, aQ1, false, lds_ld32(kr1 + 32 + hi * 16), (short)0, s1, false, false);
      s2 = __builtin_amdgcn_wmma_f32_16x16x32_bf16(false, aQ1, false, lds_ld32(kr2 + 32 + hi * 16), (short)0, s2, false, false);
      s3 = __builtin_amdgcn_wmma_f32_16x16x32_bf16(false, aQ1, false, lds_ld32(kr3 + 32 + hi * 16), (short)0, s3, false, false);
    }

    // ---- online softmax: asm row head + packed-fp32 body ------------------
    float nms[8], corrs[8];
#pragma unroll
    for (int r = 0; r < 8; ++r) {
      float mx = vmax(vmax3(s0[r], s1[r], s2[r]), s3[r]);
      mx = rowmax16(mx);                       // 4x v_max_num_f32_dpp
      const float nm = vmax(rm[r], mx);
      corrs[r] = exp2f(rm[r] - nm);
      rm[r] = nm;
      nms[r] = nm;
    }
#pragma unroll
    for (int h = 0; h < 4; ++h) {              // two rows per step, all pk ops
      const floatx2 nm2 = {nms[2 * h], nms[2 * h + 1]};
      const floatx2 c2  = {corrs[2 * h], corrs[2 * h + 1]};
      floatx2 a0 = pk_sub((floatx2){s0[2 * h], s0[2 * h + 1]}, nm2);
      floatx2 a1 = pk_sub((floatx2){s1[2 * h], s1[2 * h + 1]}, nm2);
      floatx2 a2 = pk_sub((floatx2){s2[2 * h], s2[2 * h + 1]}, nm2);
      floatx2 a3 = pk_sub((floatx2){s3[2 * h], s3[2 * h + 1]}, nm2);
      a0[0] = exp2f(a0[0]); a0[1] = exp2f(a0[1]);
      a1[0] = exp2f(a1[0]); a1[1] = exp2f(a1[1]);
      a2[0] = exp2f(a2[0]); a2[1] = exp2f(a2[1]);
      a3[0] = exp2f(a3[0]); a3[1] = exp2f(a3[1]);
      s0[2 * h] = a0[0]; s0[2 * h + 1] = a0[1];
      s1[2 * h] = a1[0]; s1[2 * h + 1] = a1[1];
      s2[2 * h] = a2[0]; s2[2 * h + 1] = a2[1];
      s3[2 * h] = a3[0]; s3[2 * h + 1] = a3[1];
      const floatx2 sum = pk_add(pk_add(a0, a1), pk_add(a2, a3));
      floatx2 rl2 = pk_fma((floatx2){rlv[2 * h], rlv[2 * h + 1]}, c2, sum);
      rlv[2 * h] = rl2[0]; rlv[2 * h + 1] = rl2[1];
      floatx2 t;
      t = pk_mul((floatx2){o0[2 * h], o0[2 * h + 1]}, c2); o0[2 * h] = t[0]; o0[2 * h + 1] = t[1];
      t = pk_mul((floatx2){o1[2 * h], o1[2 * h + 1]}, c2); o1[2 * h] = t[0]; o1[2 * h + 1] = t[1];
      t = pk_mul((floatx2){o2[2 * h], o2[2 * h + 1]}, c2); o2[2 * h] = t[0]; o2[2 * h + 1] = t[1];
      t = pk_mul((floatx2){o3[2 * h], o3[2 * h + 1]}, c2); o3[2 * h] = t[0]; o3[2 * h + 1] = t[1];
    }

    // ---- P -> per-wave LDS [q][kperm], packed pair stores -----------------
#pragma unroll
    for (int r = 0; r < 8; ++r) {
      const int q = r + 8 * hi;
      bf16x2 a; a[0] = (__bf16)s0[r]; a[1] = (__bf16)s1[r];  // kperm 2lm, 2lm+1
      bf16x2 c; c[0] = (__bf16)s2[r]; c[1] = (__bf16)s3[r];
      *(bf16x2*)&Pw[q * PSTR + 2 * lm]      = a;
      *(bf16x2*)&Pw[q * PSTR + 32 + 2 * lm] = c;
    }
    asm volatile("s_wait_dscnt 0" ::: "memory");    // cross-lane LDS RAW, same wave

    // ---- O += P * V  (2 k-chunks of 32 x 4 d-tiles of 16) -----------------
#pragma unroll
    for (int c = 0; c < 2; ++c) {
      bf16x16 pa;
      {
        const __bf16* pr = Pw + lm * PSTR + c * 32;
        bf16x8 lo = *(const bf16x8*)(pr + hi * 16);
        bf16x8 hh = *(const bf16x8*)(pr + hi * 16 + 8);
#pragma unroll
        for (int j = 0; j < 8; ++j) { pa[j] = lo[j]; pa[8 + j] = hh[j]; }
      }
      o0 = __builtin_amdgcn_wmma_f32_16x16x32_bf16(false, pa, false, lds_ld32(&VshT[( 0 + lm) * VSTR] + c * 32 + hi * 16), (short)0, o0, false, false);
      o1 = __builtin_amdgcn_wmma_f32_16x16x32_bf16(false, pa, false, lds_ld32(&VshT[(16 + lm) * VSTR] + c * 32 + hi * 16), (short)0, o1, false, false);
      o2 = __builtin_amdgcn_wmma_f32_16x16x32_bf16(false, pa, false, lds_ld32(&VshT[(32 + lm) * VSTR] + c * 32 + hi * 16), (short)0, o2, false, false);
      o3 = __builtin_amdgcn_wmma_f32_16x16x32_bf16(false, pa, false, lds_ld32(&VshT[(48 + lm) * VSTR] + c * 32 + hi * 16), (short)0, o3, false, false);
    }

    __syncthreads();   // protect Ksh/VshT before next overwrite
  }

  // ---- epilogue: reduce partial sums, v_rcp normalize, store --------------
#pragma unroll
  for (int r = 0; r < 8; ++r) {
    const float inv = __builtin_amdgcn_rcpf(rowsum16(rlv[r]));  // v_rcp_f32
    float* op = Ob + (long)(q0 + r + 8 * hi) * D_DIM + lm;
    op[0]  = o0[r] * inv;
    op[16] = o1[r] * inv;
    op[32] = o2[r] * inv;
    op[48] = o3[r] * inv;
  }
}

extern "C" void kernel_launch(void* const* d_in, const int* in_sizes, int n_in,
                              void* d_out, int out_size, void* d_ws, size_t ws_size,
                              hipStream_t stream) {
  const float* x1 = (const float*)d_in[0];  // Q
  const float* x2 = (const float*)d_in[1];  // K
  const float* x3 = (const float*)d_in[2];  // V
  float* out = (float*)d_out;
  (void)in_sizes; (void)n_in; (void)out_size; (void)d_ws; (void)ws_size;

  const int blocks = 2 * (S_LEN / 128);     // 128 WGs x 256 threads (8 waves)
  fa_fwd_kernel<<<blocks, 256, 0, stream>>>(x1, x2, x3, out);
}
